// PointsDecoder_61392262529386
// MI455X (gfx1250) — compile-verified, matrix operand order checked by reading, not run on hardware
//
#include <hip/hip_runtime.h>
#include <hip/hip_bf16.h>
#include <math.h>

#define MQ 131072
#define NP 1024

typedef __attribute__((ext_vector_type(16))) _Float16 v16h;
typedef __attribute__((ext_vector_type(8)))  float    v8f;

// ---------------------------------------------------------------------------
// Weight conversion: f32 row-major (fin x fout) -> f16 WMMA B-fragments.
// Per 32x16 (KxN) tile: lane l holds column n = l%16; K order per lane:
//   lanes 0-15 : K = {kt*32 + 0..7, kt*32 + 16..23}
//   lanes 16-31: K = {kt*32 + 8..15, kt*32 + 24..31}
// stored as 16 contiguous halves per lane -> one aligned v16h load.
// ---------------------------------------------------------------------------
__global__ void pd_conv_weights(const float* __restrict__ W, _Float16* __restrict__ dst,
                                int fin, int fout, int KT, int NT)
{
    int t = blockIdx.x * blockDim.x + threadIdx.x;
    int total = KT * NT * 512;
    if (t >= total) return;
    int j    = t & 15;
    int lane = (t >> 4) & 31;
    int tile = t >> 9;
    int nt = tile % NT;
    int kt = tile / NT;
    int half = lane >> 4;
    int grp  = j >> 3;
    int w    = j & 7;
    int k    = kt * 32 + grp * 16 + half * 8 + w;
    int ncol = nt * 16 + (lane & 15);
    float v = (k < fin && ncol < fout) ? W[(size_t)k * fout + ncol] : 0.0f;
    dst[t] = (_Float16)v;
}

// ---------------------------------------------------------------------------
// KNN: one thread per (n,m); top-4 over P=1024 points; deterministic partial
// sums of 1/dist per (n,k) (reference normalizes weights over the whole M axis)
// ---------------------------------------------------------------------------
__global__ __launch_bounds__(256) void pd_knn(const float* __restrict__ coords,
                                              const float* __restrict__ pts,
                                              int* __restrict__ idxOut,
                                              float* __restrict__ distOut,
                                              float* __restrict__ partial)
{
    __shared__ float sp[NP][4];
    __shared__ float red[256];
    int n   = blockIdx.y;
    int tid = threadIdx.x;
    int m   = blockIdx.x * 256 + tid;

    for (int i = tid; i < NP; i += 256) {
        float x = pts[((size_t)n * NP + i) * 3 + 0];
        float y = pts[((size_t)n * NP + i) * 3 + 1];
        float z = pts[((size_t)n * NP + i) * 3 + 2];
        sp[i][0] = x; sp[i][1] = y; sp[i][2] = z;
        sp[i][3] = x * x + y * y + z * z;
    }
    __syncthreads();

    float cx = coords[((size_t)n * MQ + m) * 3 + 0];
    float cy = coords[((size_t)n * MQ + m) * 3 + 1];
    float cz = coords[((size_t)n * MQ + m) * 3 + 2];
    float a2 = cx * cx + cy * cy + cz * cz;

    float bd0 = 3e38f, bd1 = 3e38f, bd2 = 3e38f, bd3 = 3e38f;
    int   bi0 = 0, bi1 = 0, bi2 = 0, bi3 = 0;
    for (int p = 0; p < NP; ++p) {
        float d2 = a2 + sp[p][3] - 2.0f * (cx * sp[p][0] + cy * sp[p][1] + cz * sp[p][2]);
        if (d2 < bd3) {
            bd3 = d2; bi3 = p;
            if (bd3 < bd2) { float t = bd2; bd2 = bd3; bd3 = t; int u = bi2; bi2 = bi3; bi3 = u; }
            if (bd2 < bd1) { float t = bd1; bd1 = bd2; bd2 = t; int u = bi1; bi1 = bi2; bi2 = u; }
            if (bd1 < bd0) { float t = bd0; bd0 = bd1; bd1 = t; int u = bi0; bi0 = bi1; bi1 = u; }
        }
    }
    size_t o = ((size_t)n * MQ + m) * 4;
    idxOut[o + 0] = bi0; idxOut[o + 1] = bi1; idxOut[o + 2] = bi2; idxOut[o + 3] = bi3;
    distOut[o + 0] = bd0; distOut[o + 1] = bd1; distOut[o + 2] = bd2; distOut[o + 3] = bd3;

    float wv[4] = {1.0f / bd0, 1.0f / bd1, 1.0f / bd2, 1.0f / bd3};
    for (int k = 0; k < 4; ++k) {
        red[tid] = wv[k];
        __syncthreads();
        for (int s = 128; s > 0; s >>= 1) {
            if (tid < s) red[tid] += red[tid + s];
            __syncthreads();
        }
        if (tid == 0) partial[((size_t)n * gridDim.x + blockIdx.x) * 4 + k] = red[0];
        __syncthreads();
    }
}

__global__ void pd_reduce_wsum(const float* __restrict__ partial, float* __restrict__ wsum,
                               int nblocks)
{
    int t = threadIdx.x;
    if (t < 8) {
        int n = t >> 2, k = t & 3;
        float s = 0.0f;
        for (int b = 0; b < nblocks; ++b) s += partial[((size_t)n * nblocks + b) * 4 + k];
        wsum[t] = s;
    }
}

// ---------------------------------------------------------------------------
// WMMA helpers (wave32, 16x16x32 f16 -> f32)
// ---------------------------------------------------------------------------
__device__ __forceinline__ v16h pd_fragA(_Float16 (*buf)[160], int kbase)
{
    int lane = threadIdx.x & 31;
    int m    = lane & 15;
    int half = lane >> 4;
    const _Float16* p = &buf[m][kbase + half * 8];
    union { v16h v; unsigned u[8]; } r;
    const unsigned* lo = (const unsigned*)p;
    const unsigned* hi = (const unsigned*)(p + 16);
#pragma unroll
    for (int i = 0; i < 4; ++i) { r.u[i] = lo[i]; r.u[4 + i] = hi[i]; }
    return r.v;
}

__device__ __forceinline__ v16h pd_fragB(const _Float16* base, int tileIdx)
{
    int lane = threadIdx.x & 31;
    return *(const v16h*)(base + ((size_t)tileIdx * 32 + lane) * 16);
}

__device__ __forceinline__ void pd_gemm(_Float16 (*in)[160], _Float16 (*out)[160],
                                        const _Float16* wf, const float* __restrict__ bias,
                                        int KT, int NT, bool relu)
{
    int lane = threadIdx.x & 31;
    int half = lane >> 4;
    int coll = lane & 15;
    for (int nt = 0; nt < NT; ++nt) {
        v8f c = {};
        for (int kt = 0; kt < KT; ++kt) {
            v16h a = pd_fragA(in, kt * 32);
            v16h b = pd_fragB(wf, kt * NT + nt);
            c = __builtin_amdgcn_wmma_f32_16x16x32_f16(false, a, false, b, (short)0, c, false, false);
        }
        int col = nt * 16 + coll;
        float bb = bias[col];
#pragma unroll
        for (int r = 0; r < 8; ++r) {
            float v = c[r] + bb;
            if (relu) v = fmaxf(v, 0.0f);
            out[r + half * 8][col] = (_Float16)v;
        }
    }
}

__device__ __forceinline__ float pd_tap(const float* __restrict__ pb, int xi, int yi)
{
    bool ok = (xi >= 0) & (xi < 256) & (yi >= 0) & (yi < 256);
    int xc = xi < 0 ? 0 : (xi > 255 ? 255 : xi);
    int yc = yi < 0 ? 0 : (yi > 255 ? 255 : yi);
    float t = pb[yc * 256 + xc];
    return ok ? t : 0.0f;
}

// harmonic-27 element for lane: [sin(e0..e11), cos(e0..e11), x0,x1,x2, 0,0,0,0,0]
__device__ __forceinline__ float pd_harm27(float rx, float ry, float rz, int lane)
{
    if (lane < 24) {
        int l = (lane < 12) ? lane : lane - 12;
        int d = l >> 2;
        float x = (d == 0) ? rx : ((d == 1) ? ry : rz);
        float e = x * (float)(1 << (l & 3));
        return (lane < 12) ? sinf(e) : cosf(e);
    } else if (lane < 27) {
        return (lane == 24) ? rx : ((lane == 25) ? ry : rz);
    }
    return 0.0f;
}

// ---------------------------------------------------------------------------
// Main decoder: 1 wave = 16 query points, all linear layers via WMMA
// ---------------------------------------------------------------------------
__global__ __launch_bounds__(128) void pd_decoder(
    const float* __restrict__ coords, const float* __restrict__ dirs,
    const float* __restrict__ pts, const float* __restrict__ tplanes,
    const float* __restrict__ pemb,
    const _Float16* __restrict__ fq1, const _Float16* __restrict__ fq2,
    const _Float16* __restrict__ ff1, const _Float16* __restrict__ ff2,
    const _Float16* __restrict__ ff3, const _Float16* __restrict__ fr1,
    const float* __restrict__ qb1, const float* __restrict__ qb2,
    const float* __restrict__ fb1, const float* __restrict__ fb2,
    const float* __restrict__ fb3,
    const float* __restrict__ dw, const float* __restrict__ db,
    const float* __restrict__ rb1,
    const float* __restrict__ rw2, const float* __restrict__ rb2,
    const float* __restrict__ wsum, const int* __restrict__ idxArr,
    const float* __restrict__ distArr,
    float* __restrict__ outDen, float* __restrict__ outRgb)
{
    __shared__ __align__(16) _Float16 sX[4][16][160];
    __shared__ __align__(16) _Float16 sY[4][16][160];
    __shared__ _Float16 sTex[4][16][32];
    __shared__ float sC[4][16][3];
    __shared__ float sD[4][16][3];
    __shared__ float sW[4][16][4];
    __shared__ float sSel[4][16];
    __shared__ int   sId[4][16];

    const int wave = threadIdx.x >> 5;
    const int lane = threadIdx.x & 31;
    const int tile = blockIdx.x * 4 + wave;        // 16384 tiles total
    const int n    = tile >> 13;                    // / (M/16 = 8192)
    const int m0   = (tile & 8191) << 4;

    _Float16 (*bx)[160] = sX[wave];
    _Float16 (*by)[160] = sY[wave];

    // ---- load coords/dirs ----
    for (int i = lane; i < 48; i += 32) {
        int p = i / 3, d = i % 3;
        sC[wave][p][d] = coords[((size_t)n * MQ + m0 + p) * 3 + d];
        sD[wave][p][d] = dirs  [((size_t)n * MQ + m0 + p) * 3 + d];
    }
    __syncthreads();
    if (lane < 16) {
        float x = sC[wave][lane][0], y = sC[wave][lane][1], z = sC[wave][lane][2];
        bool s = (x > -1.f) && (x < 1.f) && (y > -1.f) && (y < 1.f) && (z > -1.f) && (z < 1.f);
        sSel[wave][lane] = s ? 1.0f : 0.0f;
    }

    // ---- triplane bilinear sampling: lane = channel ----
    {
        const int c = lane;
        const size_t planeSz = 256 * 256;
        const float* base = tplanes + (size_t)n * 3 * 32 * planeSz;
        for (int p = 0; p < 16; ++p) {
            float cx = sC[wave][p][0], cy = sC[wave][p][1], cz = sC[wave][p][2];
            float acc = 0.0f;
#pragma unroll
            for (int pl = 0; pl < 3; ++pl) {
                float gx = (pl == 2) ? cz : cx;
                float gy = (pl == 1) ? cz : cy;
                float ix = ((gx + 1.0f) * 256.0f - 1.0f) * 0.5f;
                float iy = ((gy + 1.0f) * 256.0f - 1.0f) * 0.5f;
                float x0f = floorf(ix), y0f = floorf(iy);
                int x0 = (int)x0f, y0 = (int)y0f;
                float wx1 = ix - x0f, wx0 = 1.0f - wx1;
                float wy1 = iy - y0f, wy0 = 1.0f - wy1;
                const float* pb = base + ((size_t)pl * 32 + c) * planeSz;
                acc += pd_tap(pb, x0,     y0    ) * wx0 * wy0
                     + pd_tap(pb, x0 + 1, y0    ) * wx1 * wy0
                     + pd_tap(pb, x0,     y0 + 1) * wx0 * wy1
                     + pd_tap(pb, x0 + 1, y0 + 1) * wx1 * wy1;
            }
            sTex[wave][p][c] = (_Float16)(acc * (1.0f / 3.0f));
        }
    }

    // ---- per-neighbor q MLP, weighted accumulation in registers ----
    v8f accQ0 = {}; v8f accQ1 = {};
    for (int k = 0; k < 4; ++k) {
        if (lane < 16) {
            size_t o = ((size_t)n * MQ + m0 + lane) * 4 + k;
            int id  = idxArr[o];
            float dd = distArr[o];
            sId[wave][lane]   = id;
            sW[wave][lane][k] = (1.0f / dd) / wsum[n * 4 + k];
        }
        __syncthreads();
        // build h (16x64) in by: cols 0..31 = pts_emb, 32..58 = harmonic, 59..63 = 0
        for (int p = 0; p < 16; ++p) {
            int id = sId[wave][p];
            by[p][lane] = (_Float16)pemb[(size_t)id * 32 + lane];
            float rx = sC[wave][p][0] - pts[((size_t)n * NP + id) * 3 + 0];
            float ry = sC[wave][p][1] - pts[((size_t)n * NP + id) * 3 + 1];
            float rz = sC[wave][p][2] - pts[((size_t)n * NP + id) * 3 + 2];
            float inv = 1.0f / fmaxf(sqrtf(rx * rx + ry * ry + rz * rz), 1e-12f);
            rx *= inv; ry *= inv; rz *= inv;
            by[p][32 + lane] = (_Float16)pd_harm27(rx, ry, rz, lane);
        }
        __syncthreads();
        pd_gemm(by, bx, fq1, qb1, 2, 4, true);      // 59(pad64) -> 64, relu
        __syncthreads();
#pragma unroll
        for (int nt = 0; nt < 2; ++nt) {            // 64 -> 32 + weighted accumulate
            v8f c = {};
#pragma unroll
            for (int kt = 0; kt < 2; ++kt) {
                v16h a = pd_fragA(bx, kt * 32);
                v16h b = pd_fragB(fq2, kt * 2 + nt);
                c = __builtin_amdgcn_wmma_f32_16x16x32_f16(false, a, false, b, (short)0, c, false, false);
            }
            int col = nt * 16 + (lane & 15);
            float bb = qb2[col];
            int half = lane >> 4;
#pragma unroll
            for (int r = 0; r < 8; ++r) {
                float v = (c[r] + bb) * sW[wave][r + half * 8][k];
                if (nt == 0) accQ0[r] += v; else accQ1[r] += v;
            }
        }
        __syncthreads();
    }

    // ---- assemble feat (16x64) in bx: tex | pts_feat ----
    for (int p = 0; p < 16; ++p) bx[p][lane] = sTex[wave][p][lane];
    {
        int half = lane >> 4;
        int col  = lane & 15;
#pragma unroll
        for (int r = 0; r < 8; ++r) {
            bx[r + half * 8][32 + col]      = (_Float16)accQ0[r];
            bx[r + half * 8][48 + col]      = (_Float16)accQ1[r];
        }
    }
    __syncthreads();

    // ---- f MLP ----
    pd_gemm(bx, by, ff1, fb1, 2, 8, true);   __syncthreads();
    pd_gemm(by, bx, ff2, fb2, 4, 8, true);   __syncthreads();
    pd_gemm(bx, by, ff3, fb3, 4, 8, false);  __syncthreads();   // feat2 in by[0..127]

    // ---- density head ----
    {
        int p = lane & 15;
        float acc = db[0];
        for (int j = 0; j < 128; ++j) acc += (float)by[p][j] * dw[j];
        float z = 10.0f * acc;
        float sp = (z > 30.0f) ? z : log1pf(expf(z));
        float raw = sp * 0.1f * sSel[wave][p];
        float den = 1.0f - expf(-raw);
        if (lane < 16) outDen[(size_t)n * MQ + m0 + p] = den;
    }

    // ---- rays harmonic into by cols 128..159 ----
    for (int p = 0; p < 16; ++p) {
        float dx = sD[wave][p][0], dy = sD[wave][p][1], dz = sD[wave][p][2];
        float inv = 1.0f / fmaxf(sqrtf(dx * dx + dy * dy + dz * dz), 1e-12f);
        dx *= inv; dy *= inv; dz *= inv;
        by[p][128 + lane] = (_Float16)pd_harm27(dx, dy, dz, lane);
    }
    __syncthreads();

    // ---- r MLP: 155(pad160) -> 64, relu ----
    pd_gemm(by, bx, fr1, rb1, 5, 4, true);
    __syncthreads();

    // ---- rgb head: 64 -> 3, sigmoid scale ----
    {
        int p = lane & 15;
#pragma unroll
        for (int c3 = 0; c3 < 3; ++c3) {
            float acc = rb2[c3];
            for (int j = 0; j < 64; ++j) acc += (float)bx[p][j] * rw2[j * 3 + c3];
            float sg = 1.0f / (1.0f + expf(-acc));
            float v = sg * 1.002f - 0.001f;
            if (lane < 16) outRgb[((size_t)n * MQ + m0 + p) * 3 + c3] = v;
        }
    }
}

// ---------------------------------------------------------------------------
extern "C" void kernel_launch(void* const* d_in, const int* in_sizes, int n_in,
                              void* d_out, int out_size, void* d_ws, size_t ws_size,
                              hipStream_t stream)
{
    const float* coords = (const float*)d_in[0];
    const float* dirs   = (const float*)d_in[1];
    const float* pts    = (const float*)d_in[2];
    const float* tpl    = (const float*)d_in[3];
    const float* pemb   = (const float*)d_in[4];
    const float* qw1 = (const float*)d_in[5];  const float* qb1 = (const float*)d_in[6];
    const float* qw2 = (const float*)d_in[7];  const float* qb2 = (const float*)d_in[8];
    const float* fw1 = (const float*)d_in[9];  const float* fb1 = (const float*)d_in[10];
    const float* fw2 = (const float*)d_in[11]; const float* fb2 = (const float*)d_in[12];
    const float* fw3 = (const float*)d_in[13]; const float* fb3 = (const float*)d_in[14];
    const float* dw  = (const float*)d_in[15]; const float* db  = (const float*)d_in[16];
    const float* rw1 = (const float*)d_in[17]; const float* rb1 = (const float*)d_in[18];
    const float* rw2 = (const float*)d_in[19]; const float* rb2 = (const float*)d_in[20];

    char* ws = (char*)d_ws;
    float*     wsum    = (float*)(ws);                 // 8 floats
    float*     partial = (float*)(ws + 64);            // 2*512*4 floats
    _Float16*  wfrag   = (_Float16*)(ws + 32768);      // 57344 halves
    int*       idxArr  = (int*)(ws + 262144);          // N*M*4 ints

    float* out      = (float*)d_out;
    float* outDen   = out;                 // N*M
    float* outRgb   = out + 262144;        // N*M*3
    float* outDist  = out + 1048576;       // N*M*4

    _Float16* fq1 = wfrag;
    _Float16* fq2 = fq1 + 4096;
    _Float16* ff1 = fq2 + 2048;
    _Float16* ff2 = ff1 + 8192;
    _Float16* ff3 = ff2 + 16384;
    _Float16* fr1 = ff3 + 16384;

    auto conv = [&](const float* W, _Float16* dst, int fin, int fout, int KT, int NT) {
        int total = KT * NT * 512;
        pd_conv_weights<<<(total + 255) / 256, 256, 0, stream>>>(W, dst, fin, fout, KT, NT);
    };
    conv(qw1, fq1, 59, 64, 2, 4);
    conv(qw2, fq2, 64, 32, 2, 2);
    conv(fw1, ff1, 64, 128, 2, 8);
    conv(fw2, ff2, 128, 128, 4, 8);
    conv(fw3, ff3, 128, 128, 4, 8);
    conv(rw1, fr1, 155, 64, 5, 4);

    pd_knn<<<dim3(MQ / 256, 2), 256, 0, stream>>>(coords, pts, idxArr, outDist, partial);
    pd_reduce_wsum<<<1, 32, 0, stream>>>(partial, wsum, MQ / 256);

    pd_decoder<<<(2 * MQ / 16) / 4, 128, 0, stream>>>(
        coords, dirs, pts, tpl, pemb,
        fq1, fq2, ff1, ff2, ff3, fr1,
        qb1, qb2, fb1, fb2, fb3, dw, db, rb1, rw2, rb2,
        wsum, idxArr, outDist, outDen, outRgb);
}
